// MoEModule_17695265259900
// MI455X (gfx1250) — matrix operand order repaired and checked by reading, not compiled, not run on hardware
//
#include <hip/hip_runtime.h>

#define N_TOK 32768
#define H_DIM 512
#define O_DIM 512
#define E_NUM 16

typedef __attribute__((ext_vector_type(16))) __bf16 v16bf;
typedef __attribute__((ext_vector_type(8)))  float  v8f;
typedef unsigned int u32;
typedef unsigned short u16;
typedef __attribute__((ext_vector_type(4))) u32 u32x4;

static __device__ __forceinline__ u16 f32_to_bf16_rne(float f) {
    u32 u = __float_as_uint(f);
    u32 r = u + 0x7fffu + ((u >> 16) & 1u);
    return (u16)(r >> 16);
}

// async global -> LDS copy of 16 bytes (per lane), tracked by ASYNCcnt
static __device__ __forceinline__ void async_ld_b128(unsigned lds_off, const void* gaddr) {
    asm volatile("global_load_async_to_lds_b128 %0, %1, off"
                 :: "v"(lds_off), "v"(gaddr)
                 : "memory");
}

// ---------------- init: zero the per-expert counters ----------------
__global__ void moe_init(int* __restrict__ counts) {
    int t = threadIdx.x;
    if (t < E_NUM) counts[t] = 0;
}

// ---------------- gate: wave-per-token f32 logits + argmax ----------------
__global__ __launch_bounds__(256) void moe_gate(const float* __restrict__ x,
                                                const float* __restrict__ Wg,
                                                const float* __restrict__ bg,
                                                int* __restrict__ eidx,
                                                int* __restrict__ counts) {
    const int wave = threadIdx.x >> 5;
    const int lane = threadIdx.x & 31;
    const int n = blockIdx.x * 8 + wave;
    if (n >= N_TOK) return;

    float acc[E_NUM];
#pragma unroll
    for (int e = 0; e < E_NUM; ++e) acc[e] = 0.f;

    for (int h = lane; h < H_DIM; h += 32) {
        const float xv = x[(size_t)n * H_DIM + h];
#pragma unroll
        for (int e = 0; e < E_NUM; ++e) acc[e] += xv * Wg[e * H_DIM + h];
    }
#pragma unroll
    for (int e = 0; e < E_NUM; ++e) {
        float v = acc[e];
#pragma unroll
        for (int off = 16; off > 0; off >>= 1) v += __shfl_xor(v, off, 32);
        acc[e] = v;
    }
    if (lane == 0) {
        int best = 0;
        float bv = acc[0] + bg[0];
#pragma unroll
        for (int e = 1; e < E_NUM; ++e) {
            float lv = acc[e] + bg[e];
            if (lv > bv) { bv = lv; best = e; }   // strict > keeps first max (matches argmax)
        }
        eidx[n] = best;
        atomicAdd(&counts[best], 1);
    }
}

// ---------------- scan: 16-element exclusive prefix sum ----------------
__global__ void moe_scan(const int* __restrict__ counts,
                         int* __restrict__ seg, int* __restrict__ cursor) {
    if (threadIdx.x == 0) {
        int run = 0;
        for (int e = 0; e < E_NUM; ++e) { seg[e] = run; cursor[e] = run; run += counts[e]; }
    }
}

// ---------------- scatter tokens into per-expert segments ----------------
__global__ __launch_bounds__(256) void moe_scatter(const int* __restrict__ eidx,
                                                   int* __restrict__ cursor,
                                                   int* __restrict__ perm) {
    const int n = blockIdx.x * 256 + threadIdx.x;
    if (n >= N_TOK) return;
    const int e = eidx[n];
    const int pos = atomicAdd(&cursor[e], 1);
    perm[pos] = n;
}

// ---------------- f32 -> bf16 (RNE), 2 elems/thread, packed u32 stores ----------------
__global__ __launch_bounds__(256) void moe_cvt_bf16(const float* __restrict__ src,
                                                    u32* __restrict__ dst, int n2) {
    const int i = blockIdx.x * 256 + threadIdx.x;
    if (i >= n2) return;
    const float a = src[2 * i + 0];
    const float b = src[2 * i + 1];
    dst[i] = (u32)f32_to_bf16_rne(a) | ((u32)f32_to_bf16_rne(b) << 16);
}

// ---------------- grouped expert GEMM: bf16 WMMA, async-LDS double buffering ----------
// grid = (N_TOK/64, E_NUM, O_DIM/256), block = 256 (8 waves).
// Block tile 64x256, K-chunk 32. Wave (wm,wn) owns a 32x64 C macro-tile (2x4 WMMA tiles).
// A (64x32) and B (256x32) chunks staged in LDS via GLOBAL_LOAD_ASYNC_TO_LDS_B128,
// double-buffered with s_wait_asynccnt. LDS row stride 80B -> conflict-free ds_load_b128.
#define A_STRIDE 80
#define B_STRIDE 80
#define A_BYTES (64 * A_STRIDE)    // 5120 B per buffer
#define B_BYTES (256 * B_STRIDE)   // 20480 B per buffer

__global__ __launch_bounds__(256) void moe_gemm(const u16* __restrict__ Xb,   // [N,H] bf16
                                                const u16* __restrict__ Wb,   // [E,O,H] bf16
                                                const float* __restrict__ be, // [E,O]
                                                const int* __restrict__ counts,
                                                const int* __restrict__ seg,
                                                const int* __restrict__ perm,
                                                float* __restrict__ out) {    // [N,O]
    __shared__ __align__(16) char sm[2 * A_BYTES + 2 * B_BYTES];   // 51200 B

    const int e   = blockIdx.y;
    const int cnt = counts[e];
    const int m0  = blockIdx.x * 64;
    if (m0 >= cnt) return;                 // block-uniform early exit
    const int sbase = seg[e];

    const int tid  = threadIdx.x;
    const int lane = tid & 31;
    const int wave = tid >> 5;
    const int wm   = wave & 1;             // 2 waves along M
    const int wn   = wave >> 1;            // 4 waves along N
    const int nbb  = blockIdx.z * 256;     // block col base
    const int nb   = nbb + wn * 64;        // wave col base
    const int lr   = lane & 15;
    const bool hi  = lane >= 16;

    // --- per-thread staging assignment (chunk-invariant) ---
    // A: thread t copies 16B chunk j of row (t>>2);  B: thread t copies col t (4x16B)
    const int rowA = tid >> 2;
    const int jA   = tid & 3;
    int rA = m0 + rowA; if (rA >= cnt) rA = cnt - 1;
    const int tokA = perm[sbase + rA];
    const size_t aElem = (size_t)tokA * H_DIM;                       // ushort offset
    const size_t bElem = ((size_t)e * O_DIM + (nbb + tid)) * H_DIM;  // ushort offset

    const unsigned smbase = (unsigned)(unsigned long long)(void*)sm;
    const unsigned aBase[2] = { smbase,            smbase + A_BYTES };
    const unsigned bBase[2] = { smbase + 2 * A_BYTES, smbase + 2 * A_BYTES + B_BYTES };

    // stage one K-chunk into buffer `b` (5 async ops per wave: 1 A + 4 B)
    auto stage = [&](int kk, int b) {
        const char* gA = (const char*)(Xb + aElem + kk) + jA * 16;
        async_ld_b128(aBase[b] + (unsigned)rowA * A_STRIDE + jA * 16, gA);
        const char* gB = (const char*)(Wb + bElem + kk);
#pragma unroll
        for (int j = 0; j < 4; ++j)
            async_ld_b128(bBase[b] + (unsigned)tid * B_STRIDE + j * 16, gB + j * 16);
    };

    v8f acc[2][4];
#pragma unroll
    for (int ms = 0; ms < 2; ++ms)
#pragma unroll
        for (int ns = 0; ns < 4; ++ns)
            acc[ms][ns] = (v8f){0.f, 0.f, 0.f, 0.f, 0.f, 0.f, 0.f, 0.f};

    // per-lane K ranges within a 32-K chunk (this code's contiguous order, same for A and B):
    //   A: lane<16 -> K 0..15,       lane>=16 -> K 16..31        (two b128 at akl*2)
    //   B: lane<16 -> K 0..15,       lane>=16 -> K 16..31        (two b128 at bkl*2)
    const int aoff = hi ? 32 : 0;   // bytes
    const int boff = hi ? 32 : 0;   // bytes

    stage(0, 0);                    // prologue: chunk 0 -> buffer 0
    int cur = 0;

#pragma unroll 1
    for (int kk = 0; kk < H_DIM; kk += 32) {
        if (kk + 32 < H_DIM) {
            stage(kk + 32, cur ^ 1);                       // prefetch next chunk
            asm volatile("s_wait_asynccnt 0x5" ::: "memory");  // own chunk-kk ops done
        } else {
            asm volatile("s_wait_asynccnt 0x0" ::: "memory");
        }
        __syncthreads();            // all waves' chunk-kk data visible in LDS

        const char* bufA = sm + (cur ? A_BYTES : 0);
        const char* bufB = sm + 2 * A_BYTES + (cur ? B_BYTES : 0);

        union { v16bf v; u32x4 q[2]; } A[2], B[4];
#pragma unroll
        for (int ms = 0; ms < 2; ++ms) {
            const int row = wm * 32 + ms * 16 + lr;
            const u32x4* p = reinterpret_cast<const u32x4*>(bufA + row * A_STRIDE + aoff);
            A[ms].q[0] = p[0];      // K +0..7   (relative)
            A[ms].q[1] = p[1];      // K +8..15  (+16B)
        }
#pragma unroll
        for (int ns = 0; ns < 4; ++ns) {
            const int col = wn * 64 + ns * 16 + lr;
            const u32x4* p = reinterpret_cast<const u32x4*>(bufB + col * B_STRIDE + boff);
            B[ns].q[0] = p[0];      // 16 contiguous K values
            B[ns].q[1] = p[1];
        }
#pragma unroll
        for (int ms = 0; ms < 2; ++ms)
#pragma unroll
            for (int ns = 0; ns < 4; ++ns)
                acc[ms][ns] = __builtin_amdgcn_wmma_f32_16x16x32_bf16(
                    false, A[ms].v, false, B[ns].v, (short)0, acc[ms][ns], false, false);

        __syncthreads();            // reads done before buffer `cur` is overwritten
        cur ^= 1;
    }

    // C layout (ISA 7.12.2): VGPR v -> M = v + (hi ? 8 : 0), N = lane & 15
#pragma unroll
    for (int ms = 0; ms < 2; ++ms) {
#pragma unroll
        for (int ns = 0; ns < 4; ++ns) {
            const int col = nb + ns * 16 + lr;
            const float bias = be[e * O_DIM + col];
#pragma unroll
            for (int v = 0; v < 8; ++v) {
                const int r = m0 + wm * 32 + ms * 16 + v + (hi ? 8 : 0);
                if (r < cnt) {
                    const int t = perm[sbase + r];
                    out[(size_t)t * O_DIM + col] = acc[ms][ns][v] + bias;
                }
            }
        }
    }
}

extern "C" void kernel_launch(void* const* d_in, const int* in_sizes, int n_in,
                              void* d_out, int out_size, void* d_ws, size_t ws_size,
                              hipStream_t stream) {
    const float* x  = (const float*)d_in[0];   // [N,H]
    const float* Wg = (const float*)d_in[1];   // [E,H]
    const float* bg = (const float*)d_in[2];   // [E]
    const float* We = (const float*)d_in[3];   // [E,O,H]
    const float* be = (const float*)d_in[4];   // [E,O]
    float* out = (float*)d_out;                // [N,O]

    char* ws = (char*)d_ws;
    size_t off = 0;
    auto take = [&](size_t bytes) -> void* {
        void* p = ws + off;
        off = (off + bytes + 255) & ~(size_t)255;
        return p;
    };
    u16* xb      = (u16*)take((size_t)N_TOK * H_DIM * 2);           // 32 MB
    u16* wb      = (u16*)take((size_t)E_NUM * O_DIM * H_DIM * 2);   //  8 MB
    int* eidx    = (int*)take((size_t)N_TOK * 4);
    int* perm    = (int*)take((size_t)N_TOK * 4);
    int* counts  = (int*)take(E_NUM * 4);
    int* segs    = (int*)take(E_NUM * 4);
    int* cursor  = (int*)take(E_NUM * 4);

    moe_init<<<1, 32, 0, stream>>>(counts);

    moe_cvt_bf16<<<(N_TOK * H_DIM / 2) / 256, 256, 0, stream>>>(x, (u32*)xb, N_TOK * H_DIM / 2);
    moe_cvt_bf16<<<(E_NUM * O_DIM * H_DIM / 2) / 256, 256, 0, stream>>>(We, (u32*)wb, E_NUM * O_DIM * H_DIM / 2);

    moe_gate<<<N_TOK / 8, 256, 0, stream>>>(x, Wg, bg, eidx, counts);
    moe_scan<<<1, 32, 0, stream>>>(counts, segs, cursor);
    moe_scatter<<<N_TOK / 256, 256, 0, stream>>>(eidx, cursor, perm);

    dim3 grid(N_TOK / 64, E_NUM, O_DIM / 256);
    moe_gemm<<<grid, 256, 0, stream>>>(xb, wb, be, counts, segs, perm, out);
}
